// SpatialGraphConv_32590211842066
// MI455X (gfx1250) — compile-verified, matrix operand order checked by reading, not run on hardware
//
#include <hip/hip_runtime.h>

#define DEV __device__ __forceinline__

typedef __bf16 v16bf __attribute__((ext_vector_type(16)));
typedef float  v8f   __attribute__((ext_vector_type(8)));
typedef unsigned int v8u __attribute__((ext_vector_type(8)));

constexpr int NB   = 64;     // batch
constexpr int CIN  = 64;
constexpr int COUT = 128;
constexpr int TT   = 300;
constexpr int VV   = 25;
constexpr int NS   = 3;
constexpr int P    = TT * VV;       // 7500
constexpr int PTILES = (P + 15) / 16;   // 469

DEV unsigned short f2bf(float f) {
    unsigned u = __builtin_bit_cast(unsigned, f);
    unsigned r = u + 0x7FFFu + ((u >> 16) & 1u);
    return (unsigned short)(r >> 16);
}
DEV float bf2f(unsigned short h) {
    unsigned u = ((unsigned)h) << 16;
    return __builtin_bit_cast(float, u);
}
DEV v8f vzero() {
    v8f v;
#pragma unroll
    for (int i = 0; i < 8; i++) v[i] = 0.f;
    return v;
}
// Load one 16x16x32 bf16 A/B fragment: two 16-byte chunks at elem offsets
// (base + 8g) and (base + 8g + 16), where base = row*stride + kk*32.
DEV v16bf fragld(const unsigned short* p, int off) {
    const uint4* q0 = (const uint4*)(p + off);
    const uint4* q1 = (const uint4*)(p + off + 16);
    uint4 a = *q0, b = *q1;
    v8u u;
    u[0] = a.x; u[1] = a.y; u[2] = a.z; u[3] = a.w;
    u[4] = b.x; u[5] = b.y; u[6] = b.z; u[7] = b.w;
    return __builtin_bit_cast(v16bf, u);
}
DEV v8f wmma_bf(v16bf a, v16bf b, v8f c) {
    return __builtin_amdgcn_wmma_f32_16x16x32_bf16(false, a, false, b, (short)0, c, false, false);
}
DEV void wait_ds() { asm volatile("s_wait_dscnt 0" ::: "memory"); }

// ---------------- conversion kernels ----------------

__global__ void k_cvt(const float* __restrict__ src, unsigned short* __restrict__ dst, int n) {
    for (int i = blockIdx.x * blockDim.x + threadIdx.x; i < n; i += gridDim.x * blockDim.x)
        dst[i] = f2bf(src[i]);
}

// x (n,c,t,v) f32 -> x_ctv bf16 padded: [(n*64+c)*300+t][32], v>=25 zero
__global__ void k_xctv(const float* __restrict__ x, unsigned short* __restrict__ xctv, int total) {
    for (int i = blockIdx.x * blockDim.x + threadIdx.x; i < total; i += gridDim.x * blockDim.x) {
        int v = i & 31;
        int rest = i >> 5;          // (n*64+c)*300 + t
        unsigned short val = 0;
        if (v < VV) {
            int t = rest % TT;
            int cn = rest / TT;     // n*64+c
            val = f2bf(x[cn * P + t * VV + v]);
        }
        xctv[i] = val;
    }
}

// x (n,c,p) f32 -> x_tvc bf16: [n][p][c]  (LDS tiled transpose)
__global__ void __launch_bounds__(256) k_xtvc(const float* __restrict__ x, unsigned short* __restrict__ xtvc) {
    __shared__ __align__(16) float tl[64 * 65];
    int tid = threadIdx.x;
    int n = blockIdx.x / 118, pt = blockIdx.x % 118;
    int p0 = pt * 64;
#pragma unroll
    for (int i = 0; i < 16; i++) {
        int c = i * 4 + (tid >> 6);
        int pl = tid & 63;
        int p = p0 + pl;
        float v = (p < P) ? x[(n * 64 + c) * P + p] : 0.f;
        tl[pl * 65 + c] = v;
    }
    __syncthreads();
#pragma unroll
    for (int i = 0; i < 16; i++) {
        int pl = i * 4 + (tid >> 6);
        int c = tid & 63;
        int p = p0 + pl;
        if (p < P) xtvc[(n * P + p) * 64 + c] = f2bf(tl[pl * 65 + c]);
    }
}

// ---------------- attention kernel: scores -> softmax -> +adj -> attT ----------------
// one block (8 waves) per (n,s)
__global__ void __launch_bounds__(256) k_att(
    const unsigned short* __restrict__ xtvc,
    const unsigned short* __restrict__ waB,
    const unsigned short* __restrict__ wbB,
    const float* __restrict__ biasA, const float* __restrict__ biasB,
    const float* __restrict__ adjA, const float* __restrict__ adjB,
    unsigned short* __restrict__ attT)
{
    __shared__ __align__(16) unsigned short sAB[8][2][1024]; // per-wave A_t / B_t (32x32 bf16)
    __shared__ float sRed[1024];
    __shared__ float sAtt[1024];

    int tid = threadIdx.x;
    int wv = tid >> 5, lane = tid & 31, g = lane >> 4, m = lane & 15;
    int n = blockIdx.x / NS, s = blockIdx.x % NS;

    for (int i = tid; i < 1024; i += 256) sRed[i] = 0.f;
    __syncthreads();

    // weight fragments (t-invariant): B-side [N=i][K=c]
    v16bf WaF[2][2], WbF[2][2];
#pragma unroll
    for (int ni = 0; ni < 2; ni++)
#pragma unroll
        for (int kk = 0; kk < 2; kk++) {
            int off = (s * 32 + ni * 16 + m) * 64 + kk * 32 + g * 8;
            WaF[ni][kk] = fragld(waB, off);
            WbF[ni][kk] = fragld(wbB, off);
        }
    float bA[2], bB[2];
#pragma unroll
    for (int ni = 0; ni < 2; ni++) {
        bA[ni] = biasA[s * 32 + ni * 16 + m];
        bB[ni] = biasB[s * 32 + ni * 16 + m];
    }

    v8f accS[2][2];
#pragma unroll
    for (int a = 0; a < 2; a++)
#pragma unroll
        for (int b = 0; b < 2; b++) accS[a][b] = vzero();

    for (int t = wv; t < TT; t += 8) {
        // A-side fragments from x_tvc: rows v (clamped), K=c
        v16bf Ax[2][2];
#pragma unroll
        for (int mv = 0; mv < 2; mv++)
#pragma unroll
            for (int kk = 0; kk < 2; kk++) {
                int v_ = mv * 16 + m; if (v_ > 24) v_ = 24;
                int p = t * VV + v_;
                Ax[mv][kk] = fragld(xtvc, (n * P + p) * 64 + kk * 32 + g * 8);
            }
        v8f aA[2][2], aB[2][2];
#pragma unroll
        for (int a = 0; a < 2; a++)
#pragma unroll
            for (int b = 0; b < 2; b++) { aA[a][b] = vzero(); aB[a][b] = vzero(); }
#pragma unroll
        for (int mv = 0; mv < 2; mv++)
#pragma unroll
            for (int ni = 0; ni < 2; ni++)
#pragma unroll
                for (int kk = 0; kk < 2; kk++) {
                    aA[mv][ni] = wmma_bf(Ax[mv][kk], WaF[ni][kk], aA[mv][ni]);
                    aB[mv][ni] = wmma_bf(Ax[mv][kk], WbF[ni][kk], aB[mv][ni]);
                }
        // stage A_t/B_t (with bias) to per-wave LDS as [row v/w][32 i] bf16
        unsigned short* la = &sAB[wv][0][0];
        unsigned short* lb = &sAB[wv][1][0];
#pragma unroll
        for (int mv = 0; mv < 2; mv++)
#pragma unroll
            for (int ni = 0; ni < 2; ni++)
#pragma unroll
                for (int r = 0; r < 8; r++) {
                    int row = mv * 16 + r + g * 8;
                    int col = ni * 16 + m;
                    la[row * 32 + col] = f2bf(aA[mv][ni][r] + bA[ni]);
                    lb[row * 32 + col] = f2bf(aB[mv][ni][r] + bB[ni]);
                }
        wait_ds();   // cross-lane DS dependency within wave
        // scores += A_t^T-style product: D[v][w], K=i (32)
        v16bf Ap[2], Bp[2];
#pragma unroll
        for (int q = 0; q < 2; q++) {
            Ap[q] = fragld(la, (q * 16 + m) * 32 + g * 8);
            Bp[q] = fragld(lb, (q * 16 + m) * 32 + g * 8);
        }
#pragma unroll
        for (int mv = 0; mv < 2; mv++)
#pragma unroll
            for (int nw = 0; nw < 2; nw++)
                accS[mv][nw] = wmma_bf(Ap[mv], Bp[nw], accS[mv][nw]);
    }

    // reduce partial scores across waves
#pragma unroll
    for (int mv = 0; mv < 2; mv++)
#pragma unroll
        for (int nw = 0; nw < 2; nw++)
#pragma unroll
            for (int r = 0; r < 8; r++) {
                int v = mv * 16 + r + g * 8;
                int w = nw * 16 + m;
                atomicAdd(&sRed[v * 32 + w], accS[mv][nw][r]);
            }
    __syncthreads();

    // softmax over v (per column w), scale 1/(inter*T), then + adj
    if (tid < VV) {
        int w = tid;
        const float sc = 1.f / 9600.f;
        float mx = -1e30f;
        for (int v = 0; v < VV; v++) {
            float f = sRed[v * 32 + w] * sc;
            mx = fmaxf(mx, f);
        }
        float sum = 0.f;
        for (int v = 0; v < VV; v++) sum += __expf(sRed[v * 32 + w] * sc - mx);
        float inv = 1.f / sum;
        for (int v = 0; v < VV; v++) {
            float e = __expf(sRed[v * 32 + w] * sc - mx) * inv;
            sAtt[v * 32 + w] = e + adjA[(s * VV + v) * VV + w] + adjB[(s * VV + v) * VV + w];
        }
    }
    __syncthreads();
    // store att^T zero-padded 32x32 bf16: attT[w][v]
    if (tid < 32) {
        int w = tid;
        for (int v = 0; v < 32; v++) {
            float val = (w < VV && v < VV) ? sAtt[v * 32 + w] : 0.f;
            attT[(n * NS + s) * 1024 + w * 32 + v] = f2bf(val);
        }
    }
}

// ---------------- fused z + conv_d + down kernel ----------------
// one wave per 16-row p-tile
__global__ void __launch_bounds__(256) k_yd(
    const unsigned short* __restrict__ xctv,
    const unsigned short* __restrict__ xtvc,
    const unsigned short* __restrict__ attT,
    const unsigned short* __restrict__ wdB,
    const unsigned short* __restrict__ wdnB,
    unsigned short* __restrict__ yT,
    unsigned short* __restrict__ dT)
{
    __shared__ __align__(16) unsigned short sZ[8][1024]; // per-wave 16x64 bf16 z tile
    int tid = threadIdx.x, wv = tid >> 5, lane = tid & 31, g = lane >> 4, m = lane & 15;
    int n = blockIdx.x / 59, bq = blockIdx.x % 59;
    int tile = bq * 8 + wv;
    if (tile >= PTILES) return;   // wave-uniform, no barriers in this kernel
    int p0 = tile * 16;
    int t0 = p0 / VV;
    int t1 = (p0 + 15) / VV; if (t1 > TT - 1) t1 = TT - 1;

    v8f y[8], d[8];
#pragma unroll
    for (int i = 0; i < 8; i++) { y[i] = vzero(); d[i] = vzero(); }

    int prow = p0 + m; if (prow > P - 1) prow = P - 1;
    int wrow = prow % VV;

    for (int ss = 0; ss < NS; ss++) {
        // gathered A fragment: rows = att^T[w(p0+m)][v..]
        v16bf Aatt = fragld(attT, (n * NS + ss) * 1024 + wrow * 32 + g * 8);
        // z tiles for the (<=2) t values this p-tile spans; B = x_ctv [N=c][K=v]
        v8f az0[4], az1[4];
#pragma unroll
        for (int ct = 0; ct < 4; ct++) {
            v16bf B0 = fragld(xctv, ((n * 64 + ct * 16 + m) * TT + t0) * 32 + g * 8);
            v16bf B1 = fragld(xctv, ((n * 64 + ct * 16 + m) * TT + t1) * 32 + g * 8);
            az0[ct] = wmma_bf(Aatt, B0, vzero());
            az1[ct] = wmma_bf(Aatt, B1, vzero());
        }
        // per-row t selection, stage z tile to LDS [16 rows][64 c] bf16
        unsigned short* lz = &sZ[wv][0];
#pragma unroll
        for (int ct = 0; ct < 4; ct++)
#pragma unroll
            for (int r = 0; r < 8; r++) {
                int p = p0 + r + g * 8; if (p > P - 1) p = P - 1;
                float val = (p / VV == t0) ? az0[ct][r] : az1[ct][r];
                lz[(r + g * 8) * 64 + ct * 16 + m] = f2bf(val);
            }
        wait_ds();
        // y += z-tile @ Wd_ss^T  (A=[p][c], B=Wd [o][c])
#pragma unroll
        for (int kk = 0; kk < 2; kk++) {
            v16bf Az = fragld(lz, m * 64 + kk * 32 + g * 8);
#pragma unroll
            for (int ot = 0; ot < 8; ot++) {
                v16bf Bw = fragld(wdB, (ss * 128 + ot * 16 + m) * 64 + kk * 32 + g * 8);
                y[ot] = wmma_bf(Az, Bw, y[ot]);
            }
        }
        wait_ds();
    }
    // down path: d = x-tile @ Wdown^T
#pragma unroll
    for (int kk = 0; kk < 2; kk++) {
        v16bf Axd = fragld(xtvc, (n * P + prow) * 64 + kk * 32 + g * 8);
#pragma unroll
        for (int ot = 0; ot < 8; ot++) {
            v16bf Bd = fragld(wdnB, (ot * 16 + m) * 64 + kk * 32 + g * 8);
            d[ot] = wmma_bf(Axd, Bd, d[ot]);
        }
    }
    // store yT/dT bf16 [n][p][o]
#pragma unroll
    for (int ot = 0; ot < 8; ot++)
#pragma unroll
        for (int r = 0; r < 8; r++) {
            int p = p0 + r + g * 8;
            if (p < P) {
                int o = ot * 16 + m;
                yT[(n * P + p) * 128 + o] = f2bf(y[ot][r]);
                dT[(n * P + p) * 128 + o] = f2bf(d[ot][r]);
            }
        }
}

// ---------------- BN stats ----------------
__global__ void __launch_bounds__(256) k_stats(const unsigned short* __restrict__ yT,
                                               const unsigned short* __restrict__ dT,
                                               float* __restrict__ stats)
{
    __shared__ float red[512];
    int tid = threadIdx.x;
    for (int i = tid; i < 512; i += 256) red[i] = 0.f;
    __syncthreads();
    int o0 = (tid & 15) * 8;
    float sy[8], sy2[8], sd[8], sd2[8];
#pragma unroll
    for (int j = 0; j < 8; j++) { sy[j] = 0; sy2[j] = 0; sd[j] = 0; sd2[j] = 0; }
    const int ROWS = NB * P; // 480000
    for (int row = blockIdx.x * 16 + (tid >> 4); row < ROWS; row += 256 * 16) {
        uint4 a = *(const uint4*)(yT + row * 128 + o0);
        uint4 b = *(const uint4*)(dT + row * 128 + o0);
        unsigned uy[4] = {a.x, a.y, a.z, a.w};
        unsigned ud[4] = {b.x, b.y, b.z, b.w};
#pragma unroll
        for (int j = 0; j < 4; j++) {
            float l = bf2f((unsigned short)(uy[j] & 0xffff));
            float h = bf2f((unsigned short)(uy[j] >> 16));
            sy[2 * j] += l; sy2[2 * j] += l * l;
            sy[2 * j + 1] += h; sy2[2 * j + 1] += h * h;
            l = bf2f((unsigned short)(ud[j] & 0xffff));
            h = bf2f((unsigned short)(ud[j] >> 16));
            sd[2 * j] += l; sd2[2 * j] += l * l;
            sd[2 * j + 1] += h; sd2[2 * j + 1] += h * h;
        }
    }
#pragma unroll
    for (int j = 0; j < 8; j++) {
        atomicAdd(&red[0 + o0 + j], sy[j]);
        atomicAdd(&red[128 + o0 + j], sy2[j]);
        atomicAdd(&red[256 + o0 + j], sd[j]);
        atomicAdd(&red[384 + o0 + j], sd2[j]);
    }
    __syncthreads();
    for (int i = tid; i < 512; i += 256) atomicAdd(&stats[i], red[i]);
}

__global__ void k_coef(const float* __restrict__ stats,
                       const float* __restrict__ bnG, const float* __restrict__ bnB,
                       const float* __restrict__ dbnG, const float* __restrict__ dbnB,
                       float* __restrict__ coef)
{
    int o = threadIdx.x;
    if (o >= 128) return;
    const float inv = 1.f / (float)(NB * P);
    float my = stats[o] * inv, vy = stats[128 + o] * inv - my * my;
    float md = stats[256 + o] * inv, vd = stats[384 + o] * inv - md * md;
    float ay = bnG[o] * rsqrtf(vy + 1e-5f);
    float ad = dbnG[o] * rsqrtf(vd + 1e-5f);
    coef[o] = ay;
    coef[128 + o] = bnB[o] - ay * my;
    coef[256 + o] = ad;
    coef[384 + o] = dbnB[o] - ad * md;
}

// ---------------- epilogue: BN + BN + relu + transpose to (n,o,t,v) ----------------
__global__ void __launch_bounds__(256) k_out(const unsigned short* __restrict__ yT,
                                             const unsigned short* __restrict__ dT,
                                             const float* __restrict__ coef,
                                             float* __restrict__ out)
{
    __shared__ float so[64 * 130];
    int tid = threadIdx.x;
    int n = blockIdx.x / 118, pt = blockIdx.x % 118;
    int p0 = pt * 64;
    int oc = tid & 127;
    float c0 = coef[oc], c1 = coef[128 + oc], c2 = coef[256 + oc], c3 = coef[384 + oc];
#pragma unroll
    for (int i = 0; i < 32; i++) {
        int idx = i * 256 + tid;
        int pl = idx >> 7, o = idx & 127;   // o == oc
        int p = p0 + pl;
        float v = 0.f;
        if (p < P) {
            float yv = bf2f(yT[(n * P + p) * 128 + o]);
            float dv = bf2f(dT[(n * P + p) * 128 + o]);
            v = fmaxf(c0 * yv + c1 + c2 * dv + c3, 0.f);
        }
        so[pl * 130 + o] = v;
    }
    __syncthreads();
#pragma unroll
    for (int i = 0; i < 32; i++) {
        int idx = i * 256 + tid;
        int pl = idx & 63, o = idx >> 6;
        int p = p0 + pl;
        if (p < P) out[(long)(n * 128 + o) * P + p] = so[pl * 130 + o];
    }
}

// ---------------- host launcher ----------------
extern "C" void kernel_launch(void* const* d_in, const int* in_sizes, int n_in,
                              void* d_out, int out_size, void* d_ws, size_t ws_size,
                              hipStream_t stream)
{
    (void)in_sizes; (void)n_in; (void)out_size; (void)ws_size;

    const float* x    = (const float*)d_in[0];
    const float* adjA = (const float*)d_in[1];
    const float* adjB = (const float*)d_in[2];
    const float* waW  = (const float*)d_in[3];
    const float* baW  = (const float*)d_in[4];
    const float* wbW  = (const float*)d_in[5];
    const float* bbW  = (const float*)d_in[6];
    const float* wdW  = (const float*)d_in[7];
    /* d_in[8] conv_d_b: per-channel bias cancels under training-mode BN */
    const float* dnW  = (const float*)d_in[9];
    /* d_in[10] down_b: cancels likewise */
    const float* dbnG = (const float*)d_in[11];
    const float* dbnB = (const float*)d_in[12];
    const float* bnG  = (const float*)d_in[13];
    const float* bnB  = (const float*)d_in[14];

    char* w = (char*)d_ws;
    size_t off = 0;
    auto alloc = [&](size_t bytes) -> void* {
        void* p = w + off;
        off += (bytes + 255) & ~(size_t)255;
        return p;
    };
    unsigned short* xctv = (unsigned short*)alloc((size_t)NB * 64 * TT * 32 * 2);      // 78.6 MB
    unsigned short* xtvc = (unsigned short*)alloc((size_t)NB * P * 64 * 2 + 4096);     // 61.4 MB
    unsigned short* waB  = (unsigned short*)alloc((size_t)NS * 32 * 64 * 2);
    unsigned short* wbB  = (unsigned short*)alloc((size_t)NS * 32 * 64 * 2);
    unsigned short* wdB  = (unsigned short*)alloc((size_t)NS * 128 * 64 * 2);
    unsigned short* wdnB = (unsigned short*)alloc((size_t)128 * 64 * 2);
    unsigned short* attT = (unsigned short*)alloc((size_t)NB * NS * 1024 * 2);
    unsigned short* yT   = (unsigned short*)alloc((size_t)NB * P * 128 * 2);           // 122.9 MB
    unsigned short* dT   = (unsigned short*)alloc((size_t)NB * P * 128 * 2);           // 122.9 MB
    float* stats = (float*)alloc(512 * sizeof(float));
    float* coef  = (float*)alloc(512 * sizeof(float));

    hipMemsetAsync(stats, 0, 512 * sizeof(float), stream);

    k_cvt<<<24, 256, 0, stream>>>(waW, waB, NS * 32 * 64);
    k_cvt<<<24, 256, 0, stream>>>(wbW, wbB, NS * 32 * 64);
    k_cvt<<<96, 256, 0, stream>>>(wdW, wdB, NS * 128 * 64);
    k_cvt<<<32, 256, 0, stream>>>(dnW, wdnB, 128 * 64);
    k_xctv<<<4096, 256, 0, stream>>>(x, xctv, NB * 64 * TT * 32);
    k_xtvc<<<NB * 118, 256, 0, stream>>>(x, xtvc);
    k_att<<<NB * NS, 256, 0, stream>>>(xtvc, waB, wbB, baW, bbW, adjA, adjB, attT);
    k_yd<<<NB * 59, 256, 0, stream>>>(xctv, xtvc, attT, wdB, wdnB, yT, dT);
    k_stats<<<256, 256, 0, stream>>>(yT, dT, stats);
    k_coef<<<1, 128, 0, stream>>>(stats, bnG, bnB, dbnG, dbnB, coef);
    k_out<<<NB * 118, 256, 0, stream>>>(yT, dT, coef, (float*)d_out);
}
